// FullAttention_40372692582480
// MI455X (gfx1250) — matrix operand order repaired and checked by reading, not compiled, not run on hardware
//
#include <hip/hip_runtime.h>
#include <hip/hip_bf16.h>
#include <stdint.h>

// ---- problem constants -------------------------------------------------
#define T_SEQ   4096
#define HID     2048
#define NQH     16
#define NKVH    4
#define HD      128
#define QG_N    (NQH * HD * 2)   // 4096
#define KV_N    (NKVH * HD)      // 512
#define EPS_F   1e-6f
#define SM_SCALE 0.08838834764831845f  // 1/sqrt(128)

typedef __bf16 bf16_t;
typedef __attribute__((ext_vector_type(8)))  __bf16 bf16x8;
typedef __attribute__((ext_vector_type(16))) __bf16 bf16x16;
typedef __attribute__((ext_vector_type(8)))  float  floatx8;
typedef __attribute__((ext_vector_type(4)))  uint32_t u32x4;
typedef __attribute__((ext_vector_type(8)))  uint32_t u32x8;

#define ASPACE(n) __attribute__((address_space(n)))

union Frag16 { bf16x16 v; bf16x8 h[2]; };

__device__ __forceinline__ bf16_t f2bf(float f) {
  union { float f; uint32_t u; } a; a.f = f;
  uint32_t r = a.u + 0x7FFFu + ((a.u >> 16) & 1u);   // round-nearest-even
  union { uint16_t u; bf16_t b; } o; o.u = (uint16_t)(r >> 16);
  return o.b;
}
__device__ __forceinline__ float bf2f(bf16_t b) {
  union { uint16_t u; bf16_t b; } i; i.b = b;
  union { uint32_t u; float f; } o; o.u = ((uint32_t)i.u) << 16;
  return o.f;
}

// ---- Tensor Data Mover helpers (CDNA5 TDM, CDNA5 ISA) ----
__device__ __forceinline__ uint32_t lds_offset(const void* p) {
  // generic -> LDS addrspace pointer (32-bit) -> byte offset in LDS
  return (uint32_t)(uintptr_t)(ASPACE(3) const void*)p;
}

// 2-D tile load, bf16 elements (data_size=2B), row-padded into LDS.
// tile_x elems per row, tile_y rows; global row stride = stride_x elems;
// pad_int_code: pad every (1<<code) 8-byte units; pad_amt_code: (code+1) DWORDs.
__device__ __forceinline__ void tdm_load_2d_bf16(
    const bf16_t* gsrc, uint32_t lds_byte_addr,
    uint32_t tile_x, uint32_t tile_y,
    uint32_t tensor_x, uint32_t tensor_y,
    uint64_t stride_x, uint32_t pad_int_code, uint32_t pad_amt_code) {
  uint64_t ga = (uint64_t)(uintptr_t)gsrc;
  u32x4 g0;
  g0[0] = 1u;                                          // count=1, user D#
  g0[1] = lds_byte_addr;                               // lds_addr
  g0[2] = (uint32_t)(ga & 0xFFFFFFFFu);                // global_addr[31:0]
  g0[3] = (uint32_t)((ga >> 32) & 0x01FFFFFFu)         // global_addr[56:32]
        | (2u << 30);                                  // type=2 ("image")
  u32x8 g1;
  g1[0] = (1u << 16)                                   // data_size = 2 bytes
        | (1u << 20)                                   // pad_enable
        | (pad_int_code << 22) | (pad_amt_code << 25); // LDS row padding
  g1[1] = (tensor_x & 0xFFFFu) << 16;                  // tensor_dim0[15:0]
  g1[2] = (tensor_x >> 16) | ((tensor_y & 0xFFFFu) << 16);
  g1[3] = (tensor_y >> 16) | (tile_x << 16);           // tile_dim0
  g1[4] = tile_y & 0xFFFFu;                            // tile_dim1 (tile_dim2=0)
  g1[5] = (uint32_t)(stride_x & 0xFFFFFFFFu);          // tensor_dim0_stride
  g1[6] = (uint32_t)((stride_x >> 32) & 0xFFFFu);
  g1[7] = 0u;
  asm volatile("tensor_load_to_lds %0, %1" :: "s"(g0), "s"(g1) : "memory");
}

__device__ __forceinline__ void tdm_wait0() {
#if defined(__has_builtin) && __has_builtin(__builtin_amdgcn_s_wait_tensorcnt)
  __builtin_amdgcn_s_wait_tensorcnt(0);
#else
  asm volatile("s_wait_tensorcnt 0x0" ::: "memory");
#endif
}

// ---- fp32 -> bf16 convert ---------------------------------------------
__global__ void cvt_f32_bf16(const float* __restrict__ in,
                             bf16_t* __restrict__ out, size_t n) {
  size_t i = (size_t)blockIdx.x * blockDim.x + threadIdx.x;
  size_t stride = (size_t)gridDim.x * blockDim.x;
  for (; i < n; i += stride) out[i] = f2bf(in[i]);
}

// ---- tiled WMMA GEMM: C[M,N] = A[M,K] * B[K,N], A/B bf16 row-major ----
#define BM 128
#define BN 128
#define BK 32
#define LDA_S 40   // BK + 8 pad (elems); 80B rows = 64B data + 16B TDM pad
#define LDB_S 40   // Bt rows: [n][k]

template <bool OUT_BF16>
__global__ __launch_bounds__(256)
void gemm_bf16_wmma(const bf16_t* __restrict__ A, const bf16_t* __restrict__ B,
                    void* __restrict__ Cp, int M, int N, int K) {
  __shared__ bf16_t sA[BM * LDA_S];
  __shared__ bf16_t sBt[BN * LDB_S];   // transposed: [n][k]

  const int tid  = threadIdx.x;
  const int wave = tid >> 5, lane = tid & 31;
  const int lr = lane & 15, half = lane >> 4;
  const int wm = wave >> 2, wn = wave & 3;      // 2 x 4 waves
  const int bm = blockIdx.y * BM, bn = blockIdx.x * BN;
  const uint32_t sA_lds = lds_offset(sA);

  floatx8 acc[4][2] = {};

  for (int k0 = 0; k0 < K; k0 += BK) {
    // stage A tile 128x32 via Tensor Data Mover (row 64B -> pad 16B)
    if (wave == 0) {
      tdm_load_2d_bf16(A + (size_t)bm * K + k0, sA_lds,
                       /*tile_x=*/BK, /*tile_y=*/BM,
                       /*tensor_x=*/(uint32_t)(K - k0), /*tensor_y=*/(uint32_t)(M - bm),
                       /*stride_x=*/(uint64_t)K,
                       /*pad_int=*/3u /*64B*/, /*pad_amt=*/3u /*4 dwords*/);
    }
    // stage B tile 32x128, transposed into sBt[n][k] (all 8 waves)
#pragma unroll
    for (int i = 0; i < 2; ++i) {
      int s = tid + i * 256;
      int kk = s >> 4, ns = (s & 15) * 8;
      bf16x8 d = *(const bf16x8*)(B + (size_t)(k0 + kk) * N + bn + ns);
#pragma unroll
      for (int e = 0; e < 8; ++e) sBt[(ns + e) * LDB_S + kk] = d[e];
    }
    if (k0 + BK < K) {   // prefetch next B tile into L2
      __builtin_prefetch(B + (size_t)(k0 + BK + (tid >> 4)) * N + bn, 0, 1);
    }
    if (wave == 0) tdm_wait0();
    __syncthreads();

    Frag16 af[4], bfr[2];
#pragma unroll
    for (int mi = 0; mi < 4; ++mi) {         // A: lanes 0-15 K{0..7,16..23}
      const bf16_t* p = sA + (wm * 64 + mi * 16 + lr) * LDA_S + half * 8;
      af[mi].h[0] = *(const bf16x8*)(p);
      af[mi].h[1] = *(const bf16x8*)(p + 16);
    }
#pragma unroll
    for (int ni = 0; ni < 2; ++ni) {         // B: lanes 0-15 K{0..7,16..23}
      const bf16_t* p = sBt + (wn * 32 + ni * 16 + lr) * LDB_S + half * 8;
      bfr[ni].h[0] = *(const bf16x8*)(p);
      bfr[ni].h[1] = *(const bf16x8*)(p + 16);
    }
#pragma unroll
    for (int mi = 0; mi < 4; ++mi)
#pragma unroll
      for (int ni = 0; ni < 2; ++ni)
        acc[mi][ni] = __builtin_amdgcn_wmma_f32_16x16x32_bf16(
            false, af[mi].v, false, bfr[ni].v, (short)0, acc[mi][ni], false, false);
    __syncthreads();
  }

  // C layout: lane 0-15 -> M=vgpr, N=lane; lane 16-31 -> M=8+vgpr, N=lane-16
  const int ncol = bn + wn * 32 + lr;
  const int rbase = bm + wm * 64 + half * 8;
#pragma unroll
  for (int mi = 0; mi < 4; ++mi)
#pragma unroll
    for (int ni = 0; ni < 2; ++ni)
#pragma unroll
      for (int v = 0; v < 8; ++v) {
        size_t idx = (size_t)(rbase + mi * 16 + v) * N + (ncol + ni * 16);
        if (OUT_BF16) ((bf16_t*)Cp)[idx] = f2bf(acc[mi][ni][v]);
        else          ((float*)Cp)[idx]  = acc[mi][ni][v];
      }
}

// ---- RMSNorm + RoPE (per (t, head) block, 128 threads) -----------------
__global__ __launch_bounds__(128)
void norm_rope(const bf16_t* __restrict__ in, const float* __restrict__ w,
               const int* __restrict__ pos, const float* __restrict__ invf,
               bf16_t* __restrict__ out, int ld_in, int head_stride,
               int ld_out) {
  const int t = blockIdx.x, h = blockIdx.y, d = threadIdx.x;
  float x = bf2f(in[(size_t)t * ld_in + h * head_stride + d]);

  float ss = x * x;
#pragma unroll
  for (int off = 1; off < 32; off <<= 1) ss += __shfl_xor(ss, off, 32);
  __shared__ float red[4];
  if ((threadIdx.x & 31) == 0) red[threadIdx.x >> 5] = ss;
  __syncthreads();
  float tot = red[0] + red[1] + red[2] + red[3];
  float n = x * rsqrtf(tot * (1.0f / HD) + EPS_F) * w[d];

  __shared__ float nv[HD];
  nv[d] = n;
  __syncthreads();
  float other = nv[d ^ 64];
  float rot = (d < 64) ? -other : other;

  float p = (float)pos[t];
  float fr = invf[d & 63];
  float y = n * __cosf(p * fr) + rot * __sinf(p * fr);
  out[(size_t)t * ld_out + h * HD + d] = f2bf(y);
}

// ---- flash attention: 64 query rows/block, 4 waves x 16 rows -----------
#define BR 64
#define BC 64
#define KLD (HD + 8)   // 136 elems -> 272B rows = 256B data + 16B TDM pad
#define VLD (BC + 8)   // 72 elems  -> 144B rows
#define PLD (BC + 8)

__global__ __launch_bounds__(128)
void attn_fwd(const bf16_t* __restrict__ Q,   // [T][NQH*HD] post-RoPE
              const bf16_t* __restrict__ Kh,  // [T][NKVH*HD] post-RoPE
              const bf16_t* __restrict__ Vh,  // [T][NKVH*HD]
              const bf16_t* __restrict__ QG,  // [T][QG_N], gate at +HD
              bf16_t* __restrict__ O) {       // [T][NQH*HD]
  const int h = blockIdx.y, kvh = h >> 2;
  const int q0 = blockIdx.x * BR;
  const int tid = threadIdx.x, wave = tid >> 5, lane = tid & 31;
  const int lr = lane & 15, half = lane >> 4;

  __shared__ bf16_t sK[BC * KLD];
  __shared__ bf16_t sVt[HD * VLD];          // [d][kv]
  __shared__ bf16_t sP[4][16 * PLD];        // per-wave P tile
  const uint32_t sK_lds = lds_offset(sK);

  // Q A-fragments straight from global (rows = q0 + wave*16 + lr)
  Frag16 qf[4];
  const bf16_t* qp = Q + (size_t)(q0 + wave * 16 + lr) * (NQH * HD)
                       + h * HD + half * 8;
#pragma unroll
  for (int kk = 0; kk < 4; ++kk) {
    qf[kk].h[0] = *(const bf16x8*)(qp + kk * 32);
    qf[kk].h[1] = *(const bf16x8*)(qp + kk * 32 + 16);
  }

  floatx8 o_acc[8] = {};
  float m_i[8], l_i[8];
#pragma unroll
  for (int v = 0; v < 8; ++v) { m_i[v] = -1e30f; l_i[v] = 0.f; }

  for (int c0 = 0; c0 < q0 + BR; c0 += BC) {
    // K tile [64][128] via TDM (row 256B -> pad 16B)
    if (wave == 0) {
      tdm_load_2d_bf16(Kh + (size_t)c0 * KV_N + kvh * HD, sK_lds,
                       /*tile_x=*/HD, /*tile_y=*/BC,
                       /*tensor_x=*/HD, /*tensor_y=*/(uint32_t)(T_SEQ - c0),
                       /*stride_x=*/(uint64_t)KV_N,
                       /*pad_int=*/5u /*256B*/, /*pad_amt=*/3u /*4 dwords*/);
    }
    // V tile transposed [128][64] (manual: TDM cannot transpose)
#pragma unroll
    for (int i = 0; i < 8; ++i) {
      int s = tid + i * 128, r = s >> 3, cs = (s & 7) * 8;
      bf16x8 dv = *(const bf16x8*)(Vh + (size_t)(c0 + r) * KV_N + kvh * HD + cs);
#pragma unroll
      for (int e = 0; e < 8; ++e) sVt[(cs + e) * VLD + r] = dv[e];
    }
    if (wave == 0) tdm_wait0();
    __syncthreads();

    // S = Q K^T  (16 x 64 per wave)
    floatx8 s_acc[4] = {};
#pragma unroll
    for (int ni = 0; ni < 4; ++ni)
#pragma unroll
      for (int kk = 0; kk < 4; ++kk) {
        Frag16 bk;
        const bf16_t* p = sK + (ni * 16 + lr) * KLD + kk * 32 + half * 8;
        bk.h[0] = *(const bf16x8*)(p);
        bk.h[1] = *(const bf16x8*)(p + 16);
        s_acc[ni] = __builtin_amdgcn_wmma_f32_16x16x32_bf16(
            false, qf[kk].v, false, bk.v, (short)0, s_acc[ni], false, false);
      }

    // online softmax (rows = wave*16 + half*8 + v, cols = c0 + ni*16 + lr)
    const int rloc = q0 + wave * 16 + half * 8;
#pragma unroll
    for (int v = 0; v < 8; ++v) {
      const int row = rloc + v;
      float rowmax = -1e30f;
#pragma unroll
      for (int ni = 0; ni < 4; ++ni) {
        float sv = s_acc[ni][v] * SM_SCALE;
        if (c0 + ni * 16 + lr > row) sv = -1e30f;
        s_acc[ni][v] = sv;
        rowmax = fmaxf(rowmax, sv);
      }
#pragma unroll
      for (int off = 1; off < 16; off <<= 1)
        rowmax = fmaxf(rowmax, __shfl_xor(rowmax, off, 32));
      float mnew = fmaxf(m_i[v], rowmax);
      float corr = __expf(m_i[v] - mnew);
      float rowsum = 0.f;
#pragma unroll
      for (int ni = 0; ni < 4; ++ni) {
        float pv = __expf(s_acc[ni][v] - mnew);
        s_acc[ni][v] = pv;
        rowsum += pv;
      }
#pragma unroll
      for (int off = 1; off < 16; off <<= 1) rowsum += __shfl_xor(rowsum, off, 32);
      l_i[v] = l_i[v] * corr + rowsum;
      m_i[v] = mnew;
#pragma unroll
      for (int nf = 0; nf < 8; ++nf) o_acc[nf][v] *= corr;
    }

    // P -> LDS (reshape C-layout -> A-layout), per-wave region
#pragma unroll
    for (int ni = 0; ni < 4; ++ni)
#pragma unroll
      for (int v = 0; v < 8; ++v)
        sP[wave][(half * 8 + v) * PLD + ni * 16 + lr] = f2bf(s_acc[ni][v]);

    // O += P V
#pragma unroll
    for (int kk = 0; kk < 2; ++kk) {
      Frag16 pf;
      const bf16_t* pp = &sP[wave][lr * PLD + kk * 32 + half * 8];
      pf.h[0] = *(const bf16x8*)(pp);
      pf.h[1] = *(const bf16x8*)(pp + 16);
#pragma unroll
      for (int nf = 0; nf < 8; ++nf) {
        Frag16 bv;
        const bf16_t* vp = sVt + (nf * 16 + lr) * VLD + kk * 32 + half * 8;
        bv.h[0] = *(const bf16x8*)(vp);
        bv.h[1] = *(const bf16x8*)(vp + 16);
        o_acc[nf] = __builtin_amdgcn_wmma_f32_16x16x32_bf16(
            false, pf.v, false, bv.v, (short)0, o_acc[nf], false, false);
      }
    }
    __syncthreads();
  }

  // epilogue: 1/l, sigmoid(gate), store bf16 head-major
  const int rb = q0 + wave * 16 + half * 8;
#pragma unroll
  for (int nf = 0; nf < 8; ++nf)
#pragma unroll
    for (int v = 0; v < 8; ++v) {
      int row = rb + v, d = nf * 16 + lr;
      float g = bf2f(QG[(size_t)row * QG_N + h * (2 * HD) + HD + d]);
      float sig = 1.0f / (1.0f + __expf(-g));
      float val = o_acc[nf][v] / l_i[v] * sig;
      O[(size_t)row * (NQH * HD) + h * HD + d] = f2bf(val);
    }
}

// ---- host-side orchestration ------------------------------------------
extern "C" void kernel_launch(void* const* d_in, const int* in_sizes, int n_in,
                              void* d_out, int out_size, void* d_ws, size_t ws_size,
                              hipStream_t stream) {
  const float* x     = (const float*)d_in[0];
  const int*   pos   = (const int*)d_in[1];
  const float* invf  = (const float*)d_in[2];
  const float* Wq    = (const float*)d_in[3];
  const float* Wk    = (const float*)d_in[4];
  const float* Wv    = (const float*)d_in[5];
  const float* Wo    = (const float*)d_in[6];
  const float* qnw   = (const float*)d_in[7];
  const float* knw   = (const float*)d_in[8];

  char* ws = (char*)d_ws;
  const size_t MB = 1u << 20;
  bf16_t* xb    = (bf16_t*)(ws + 0);          // 16 MB  [T][HID]
  bf16_t* wq_b  = (bf16_t*)(ws + 16 * MB);    // 16 MB  [HID][QG_N]
  bf16_t* wk_b  = (bf16_t*)(ws + 32 * MB);    //  2 MB
  bf16_t* wv_b  = (bf16_t*)(ws + 34 * MB);    //  2 MB
  bf16_t* wo_b  = (bf16_t*)(ws + 36 * MB);    //  8 MB  [HID][HID]
  bf16_t* qg_b  = (bf16_t*)(ws + 44 * MB);    // 32 MB  [T][QG_N]
  bf16_t* kp_b  = (bf16_t*)(ws + 76 * MB);    //  4 MB  [T][KV_N]
  bf16_t* vp_b  = (bf16_t*)(ws + 80 * MB);    //  4 MB  [T][KV_N]
  bf16_t* qrot  = (bf16_t*)(ws + 84 * MB);    // 16 MB  [T][NQH*HD]
  bf16_t* krot  = (bf16_t*)(ws + 100 * MB);   //  4 MB  [T][KV_N]
  bf16_t* ao_b  = (bf16_t*)(ws + 104 * MB);   // 16 MB  [T][HID]

  // 1) fp32 -> bf16
  cvt_f32_bf16<<<2048, 256, 0, stream>>>(x,  xb,   (size_t)T_SEQ * HID);
  cvt_f32_bf16<<<2048, 256, 0, stream>>>(Wq, wq_b, (size_t)HID * QG_N);
  cvt_f32_bf16<<<512,  256, 0, stream>>>(Wk, wk_b, (size_t)HID * KV_N);
  cvt_f32_bf16<<<512,  256, 0, stream>>>(Wv, wv_b, (size_t)HID * KV_N);
  cvt_f32_bf16<<<1024, 256, 0, stream>>>(Wo, wo_b, (size_t)HID * HID);

  // 2) projections
  gemm_bf16_wmma<true><<<dim3(QG_N / BN, T_SEQ / BM), 256, 0, stream>>>(
      xb, wq_b, qg_b, T_SEQ, QG_N, HID);
  gemm_bf16_wmma<true><<<dim3(KV_N / BN, T_SEQ / BM), 256, 0, stream>>>(
      xb, wk_b, kp_b, T_SEQ, KV_N, HID);
  gemm_bf16_wmma<true><<<dim3(KV_N / BN, T_SEQ / BM), 256, 0, stream>>>(
      xb, wv_b, vp_b, T_SEQ, KV_N, HID);

  // 3) per-head RMSNorm + RoPE
  norm_rope<<<dim3(T_SEQ, NQH), 128, 0, stream>>>(qg_b, qnw, pos, invf, qrot,
                                                  QG_N, 2 * HD, NQH * HD);
  norm_rope<<<dim3(T_SEQ, NKVH), 128, 0, stream>>>(kp_b, knw, pos, invf, krot,
                                                   KV_N, HD, KV_N);

  // 4) causal flash attention + gate
  attn_fwd<<<dim3(T_SEQ / BR, NQH), 128, 0, stream>>>(qrot, krot, vp_b, qg_b, ao_b);

  // 5) output projection (fp32 out)
  gemm_bf16_wmma<false><<<dim3(HID / BN, T_SEQ / BM), 256, 0, stream>>>(
      ao_b, wo_b, d_out, T_SEQ, HID, HID);
}
